// BalancedMoE_2250562863538
// MI455X (gfx1250) — matrix operand order repaired and checked by reading, not compiled, not run on hardware
//
#include <hip/hip_runtime.h>
#include <hip/hip_bf16.h>

// ---- problem constants (match reference) ----
#define B_TOK  8192
#define D_LATD 1024
#define D_EMBD 1024
#define DIM    2048      // D_LAT + D_EMB, also output dim
#define NEXP   8

// ---- GEMM tiling ----
#define TILE_M 128
#define TILE_N 128
#define TILE_K 32        // one v_wmma_f32_16x16x32_f16 per K-step
#define LDA    (TILE_K + 8)   // 40 halfs = 80B row stride (16B multiple)

typedef __attribute__((ext_vector_type(4)))  _Float16 v4h;
typedef __attribute__((ext_vector_type(8)))  _Float16 v8h;
typedef __attribute__((ext_vector_type(16))) _Float16 v16h;
typedef __attribute__((ext_vector_type(8)))  float    v8f;
typedef __attribute__((ext_vector_type(4)))  unsigned v4u;
typedef __attribute__((ext_vector_type(4)))  int      v4i;
typedef __attribute__((ext_vector_type(8)))  int      v8i;

// generic pointer -> LDS byte offset (ISA 10.2: LDS aperture uses addr[31:0])
__device__ __forceinline__ unsigned lds_off(const void* p) {
    return (unsigned)(unsigned long long)p;
}

// ------------------------------------------------------------------
// 1) zero output + expert counters
// ------------------------------------------------------------------
__global__ void moe_init(float* __restrict__ out, int* __restrict__ counts) {
    size_t i = (size_t)blockIdx.x * blockDim.x + threadIdx.x;
    float4 z = make_float4(0.f, 0.f, 0.f, 0.f);
    ((float4*)out)[i] = z;                       // B_TOK*DIM/4 threads
    if (blockIdx.x == 0 && threadIdx.x < NEXP) counts[threadIdx.x] = 0;
}

// ------------------------------------------------------------------
// 2a) inp_h[b][d] = fp16(concat(x,y)[b][d])
// ------------------------------------------------------------------
__global__ void convert_inp(const float* __restrict__ x, const float* __restrict__ y,
                            _Float16* __restrict__ inp_h) {
    size_t i4 = (size_t)blockIdx.x * blockDim.x + threadIdx.x;  // over B*DIM/4
    int b = (int)(i4 / (DIM / 4));
    int d = (int)(i4 % (DIM / 4)) * 4;
    const float* s = (d < D_LATD) ? (x + (size_t)b * D_LATD + d)
                                  : (y + (size_t)b * D_EMBD + (d - D_LATD));
    float4 f = *(const float4*)s;
    v4h h; h[0] = (_Float16)f.x; h[1] = (_Float16)f.y;
           h[2] = (_Float16)f.z; h[3] = (_Float16)f.w;
    *(v4h*)(inp_h + i4 * 4) = h;
}

// ------------------------------------------------------------------
// 2b) W_h = fp16(W_experts)   (same [E][O][D] layout, contiguous in D=K)
// ------------------------------------------------------------------
__global__ void convert_w(const float* __restrict__ W, _Float16* __restrict__ Wh) {
    size_t i4 = (size_t)blockIdx.x * blockDim.x + threadIdx.x;  // over E*DIM*DIM/4
    float4 f = *(const float4*)(W + i4 * 4);
    v4h h; h[0] = (_Float16)f.x; h[1] = (_Float16)f.y;
           h[2] = (_Float16)f.z; h[3] = (_Float16)f.w;
    *(v4h*)(Wh + i4 * 4) = h;
}

// ------------------------------------------------------------------
// 3) gate logits -> top-2 -> softmax(2) -> append to per-expert lists
// ------------------------------------------------------------------
__global__ void gate_route(const float* __restrict__ x, const float* __restrict__ y,
                           const float* __restrict__ Wg, const float* __restrict__ bg,
                           int* __restrict__ counts, int* __restrict__ tok,
                           float* __restrict__ gw) {
    int t = blockIdx.x * blockDim.x + threadIdx.x;
    if (t >= B_TOK) return;
    float acc[NEXP];
#pragma unroll
    for (int e = 0; e < NEXP; ++e) acc[e] = bg[e];
    const float* xr = x + (size_t)t * D_LATD;
    const float* yr = y + (size_t)t * D_EMBD;
    for (int d = 0; d < DIM; ++d) {
        float v = (d < D_LATD) ? xr[d] : yr[d - D_LATD];
#pragma unroll
        for (int e = 0; e < NEXP; ++e) acc[e] += v * Wg[e * DIM + d];
    }
    int i0 = 0;
#pragma unroll
    for (int e = 1; e < NEXP; ++e) if (acc[e] > acc[i0]) i0 = e;
    int i1 = -1;
#pragma unroll
    for (int e = 0; e < NEXP; ++e)
        if (e != i0 && (i1 < 0 || acc[e] > acc[i1])) i1 = e;
    float g0 = 1.f / (1.f + __expf(acc[i1] - acc[i0]));
    float g1 = 1.f - g0;
    int p0 = atomicAdd(&counts[i0], 1);
    tok[i0 * B_TOK + p0] = t;  gw[i0 * B_TOK + p0] = g0;
    int p1 = atomicAdd(&counts[i1], 1);
    tok[i1 * B_TOK + p1] = t;  gw[i1 * B_TOK + p1] = g1;
}

// ------------------------------------------------------------------
// 4) grouped expert GEMM, CDNA5 data-movement:
//    A tile : global_load_async_to_lds_b128 (token-gathered, ASYNCcnt)
//    B tile : TDM tensor_load_to_lds, 2-D D# (dense W tile, TENSORcnt)
//    B frags: plain LDS loads from Bs[n][k] (k already contiguous)
// ------------------------------------------------------------------
__global__ __launch_bounds__(256) void moe_gemm(
    const _Float16* __restrict__ inp_h, const _Float16* __restrict__ Wh,
    const float* __restrict__ bias,
    const int* __restrict__ counts, const int* __restrict__ tok,
    const float* __restrict__ gw, float* __restrict__ out) {

    const int tilesM = B_TOK / TILE_M;   // 64
    const int tilesN = DIM / TILE_N;     // 16
    int bid = blockIdx.x;
    int e   = bid / (tilesM * tilesN);
    int rem = bid % (tilesM * tilesN);
    int m0  = (rem / tilesN) * TILE_M;
    int n0  = (rem % tilesN) * TILE_N;

    int cnt = counts[e];
    if (m0 >= cnt) return;               // tile has no routed tokens

    __shared__ _Float16 As[TILE_M][LDA];      // gathered tokens, [m][k], padded
    __shared__ _Float16 Bs[TILE_N][TILE_K];   // W tile, [n][k], contiguous 8KB

    int tid  = threadIdx.x;
    int lane = tid & 31;
    int wave = tid >> 5;                 // 0..7
    int wm   = (wave & 3) * 32;          // wave m-offset in tile
    int wn   = (wave >> 2) * 64;         // wave n-offset in tile

    // per-thread A staging: thread -> (row, 16-half chunk)
    int aRow  = tid >> 1;                // 0..127
    int aHalf = (tid & 1) * 16;          // 0 or 16 within K-step
    int aIdx  = m0 + aRow;
    int aClamp = (aIdx < cnt) ? aIdx : (cnt - 1);   // OOR rows feed discarded C rows
    int aTok  = tok[e * B_TOK + aClamp];
    const _Float16* aPtr = inp_h + (size_t)aTok * DIM;
    unsigned ldsA0 = lds_off(&As[aRow][aHalf]);
    unsigned ldsA1 = ldsA0 + 16;

    // TDM descriptor (wave 0 moves the whole 128x32 B tile per K-step)
    unsigned ldsB = lds_off(&Bs[0][0]);
    unsigned long long wAddr0 =
        (unsigned long long)(const void*)(Wh + ((size_t)e * DIM + n0) * DIM);
    v8i g1;                              // D# group 1 (ISA 8.4), 2-D tensor
    g1[0] = 0x00010000;                  // wg_mask=0, data_size=2B, no flags
    g1[1] = (int)((DIM & 0xFFFF) << 16);            // tensor_dim0[15:0]
    g1[2] = (int)(((unsigned)DIM >> 16) | ((DIM & 0xFFFF) << 16)); // dim0 hi | dim1 lo
    g1[3] = (int)(((unsigned)DIM >> 16) | (TILE_K << 16));         // dim1 hi | tile_dim0
    g1[4] = TILE_N;                      // tile_dim1 = 128 rows (tile_dim2 = 0)
    g1[5] = DIM;                         // tensor_dim0_stride lo32 = 2048
    g1[6] = 0;
    g1[7] = 0;
    v4i z4; z4[0] = z4[1] = z4[2] = z4[3] = 0;      // groups 2/3 unused (2-D)

    v8f acc[2][4];
#pragma unroll
    for (int mi = 0; mi < 2; ++mi)
#pragma unroll
        for (int ni = 0; ni < 4; ++ni)
#pragma unroll
            for (int j = 0; j < 8; ++j) acc[mi][ni][j] = 0.f;

    const int half = lane >> 4;          // which 16-lane half
    const int ml   = lane & 15;

    for (int k0 = 0; k0 < DIM; k0 += TILE_K) {
        // ---- A tile: per-lane async copy global -> LDS (ASYNCcnt) ----
        {
            const _Float16* s = aPtr + k0 + aHalf;
            asm volatile("global_load_async_to_lds_b128 %0, %1, off"
                         :: "v"(ldsA0), "v"(s) : "memory");
            asm volatile("global_load_async_to_lds_b128 %0, %1, off"
                         :: "v"(ldsA1), "v"(s + 8) : "memory");
            if (k0 + TILE_K < DIM)
                __builtin_prefetch(s + TILE_K, 0, 0);   // global_prefetch_b8
        }
        // ---- B tile: one TDM descriptor moves 128x32 of W (TENSORcnt) ----
        if (wave == 0) {
            unsigned long long wa = wAddr0 + (unsigned)k0 * 2u;
            v4u g0;
            g0[0] = 1u;                              // count=1, user D#
            g0[1] = ldsB;                            // LDS byte address
            g0[2] = (unsigned)(wa & 0xFFFFFFFFu);    // global_addr[31:0]
            g0[3] = (unsigned)((wa >> 32) & 0x1FFFFFFu) | (2u << 30); // addr hi | type=2
#if __clang_major__ >= 23
            v8i z8; z8[0]=z8[1]=z8[2]=z8[3]=z8[4]=z8[5]=z8[6]=z8[7]=0;
            __builtin_amdgcn_tensor_load_to_lds(g0, g1, z4, z4, z8, 0);
#else
            __builtin_amdgcn_tensor_load_to_lds(g0, g1, z4, z4, 0);
#endif
            __builtin_amdgcn_s_wait_tensorcnt(0);
        }
#if __has_builtin(__builtin_amdgcn_s_wait_asynccnt)
        __builtin_amdgcn_s_wait_asynccnt(0);
#else
        asm volatile("s_wait_asynccnt 0x0" ::: "memory");
#endif
        __syncthreads();

        // ---- A fragments (ISA 7.12.2 16-bit A layout), plain LDS loads ----
        v16h af[2];
#pragma unroll
        for (int mi = 0; mi < 2; ++mi) {
            const _Float16* rp = &As[wm + mi * 16 + ml][0];
            v8h lo = *(const v8h*)(rp + half * 8);        // K 0-7  / 8-15
            v8h hi = *(const v8h*)(rp + 16 + half * 8);   // K 16-23 / 24-31
            af[mi] = __builtin_shufflevector(lo, hi, 0, 1, 2, 3, 4, 5, 6, 7,
                                             8, 9, 10, 11, 12, 13, 14, 15);
        }
        // ---- B fragments via plain LDS loads (2 x 8 halfs per frag) ----
        v8h bl[4], bh[4];
#pragma unroll
        for (int ni = 0; ni < 4; ++ni) {
            // lane -> column n=ml of Bs[n][k]; K 0-7/8-15 and K 16-23/24-31 by half
            const _Float16* bp = &Bs[wn + ni * 16 + ml][0];
            bl[ni] = *(const v8h*)(bp + half * 8);
            bh[ni] = *(const v8h*)(bp + 16 + half * 8);
        }
        // fence: tie all B-fragment LDS load results to a DScnt wait so WMMAs can't hoist
        asm volatile("s_wait_dscnt 0x0"
                     : "+v"(bl[0]), "+v"(bl[1]), "+v"(bl[2]), "+v"(bl[3]),
                       "+v"(bh[0]), "+v"(bh[1]), "+v"(bh[2]), "+v"(bh[3])
                     :: "memory");
        v16h bf[4];
#pragma unroll
        for (int ni = 0; ni < 4; ++ni)
            bf[ni] = __builtin_shufflevector(bl[ni], bh[ni], 0, 1, 2, 3, 4, 5, 6,
                                             7, 8, 9, 10, 11, 12, 13, 14, 15);
#pragma unroll
        for (int mi = 0; mi < 2; ++mi)
#pragma unroll
            for (int ni = 0; ni < 4; ++ni)
                acc[mi][ni] = __builtin_amdgcn_wmma_f32_16x16x32_f16(
                    false, af[mi], false, bf[ni], (short)0, acc[mi][ni],
                    false, false);
        __syncthreads();
    }

    // ---- epilogue: +bias, *gate, scatter-accumulate (2 commutative fp adds
    //      per output element -> deterministic) ----
#pragma unroll
    for (int mi = 0; mi < 2; ++mi) {
#pragma unroll
        for (int ni = 0; ni < 4; ++ni) {
            int col = n0 + wn + ni * 16 + ml;
            float bv = bias[e * DIM + col];
#pragma unroll
            for (int r = 0; r < 8; ++r) {
                int row = m0 + wm + mi * 16 + half * 8 + r;  // lanes 16-31: M+8
                if (row < cnt) {
                    int   t = tok[e * B_TOK + row];
                    float g = gw[e * B_TOK + row];
                    atomicAdd(&out[(size_t)t * DIM + col],
                              g * (acc[mi][ni][r] + bv));
                }
            }
        }
    }
}

// ------------------------------------------------------------------
extern "C" void kernel_launch(void* const* d_in, const int* in_sizes, int n_in,
                              void* d_out, int out_size, void* d_ws, size_t ws_size,
                              hipStream_t stream) {
    const float* x  = (const float*)d_in[0];   // [B, 1024]
    const float* y  = (const float*)d_in[1];   // [B, 1024]
    const float* We = (const float*)d_in[2];   // [E, D, D]
    const float* be = (const float*)d_in[3];   // [E, D]
    const float* Wg = (const float*)d_in[4];   // [E, D]
    const float* bg = (const float*)d_in[5];   // [E]
    float* out = (float*)d_out;                // [B, D]

    // workspace carve-up (~101 MB total)
    char* p = (char*)d_ws;
    _Float16* inp_h = (_Float16*)p; p += (size_t)B_TOK * DIM * sizeof(_Float16);
    _Float16* Wh    = (_Float16*)p; p += (size_t)NEXP * DIM * DIM * sizeof(_Float16);
    int*   counts = (int*)p;   p += 256;
    int*   tok    = (int*)p;   p += (size_t)NEXP * B_TOK * sizeof(int);
    float* gw     = (float*)p; p += (size_t)NEXP * B_TOK * sizeof(float);

    moe_init<<<(B_TOK * DIM / 4) / 256, 256, 0, stream>>>(out, counts);
    convert_inp<<<(B_TOK * DIM / 4) / 256, 256, 0, stream>>>(x, y, inp_h);
    convert_w<<<((size_t)NEXP * DIM * DIM / 4) / 256, 256, 0, stream>>>(We, Wh);
    gate_route<<<B_TOK / 256, 256, 0, stream>>>(x, y, Wg, bg, counts, tok, gw);
    int blocks = NEXP * (B_TOK / TILE_M) * (DIM / TILE_N);
    moe_gemm<<<blocks, 256, 0, stream>>>(inp_h, Wh, be, counts, tok, gw, out);
}
